// MultiTimeframeLNN_17575006175780
// MI455X (gfx1250) — matrix operand order repaired and checked by reading, not compiled, not run on hardware
//
#include <hip/hip_runtime.h>
#include <hip/hip_bf16.h>
#include <math.h>

typedef __bf16 bf16;
typedef __bf16 v16bf __attribute__((ext_vector_type(16)));
typedef __bf16 v8bf  __attribute__((ext_vector_type(8)));
typedef float  v8f   __attribute__((ext_vector_type(8)));
typedef float  f4    __attribute__((ext_vector_type(4)));
typedef unsigned int u32x4 __attribute__((ext_vector_type(4)));
typedef int i32x4 __attribute__((ext_vector_type(4)));
typedef int i32x8 __attribute__((ext_vector_type(8)));

#define NTF 6
#define BT 32          // batch tile per workgroup
#define THREADS 256    // 8 waves of 32

#if defined(__HIP_DEVICE_COMPILE__) && __has_builtin(__builtin_amdgcn_tensor_load_to_lds) && __has_builtin(__builtin_amdgcn_s_wait_tensorcnt)
#define USE_TDM 1
#endif

// ---------------------------------------------------------------------------
// Branch-free fast transcendentals (v_exp_f32 / v_rcp_f32 / v_rsq_f32 are
// single TRANS32 ops on gfx1250 and co-execute with WMMA per ISA 8.4).
// ---------------------------------------------------------------------------
#define LOG2E 1.4426950408889634f
__device__ __forceinline__ float fexp2_(float x) {
#if defined(__HIP_DEVICE_COMPILE__)
  return __builtin_amdgcn_exp2f(x);
#else
  return exp2f(x);
#endif
}
__device__ __forceinline__ float frcp_(float x) {
#if defined(__HIP_DEVICE_COMPILE__)
  return __builtin_amdgcn_rcpf(x);
#else
  return 1.0f / x;
#endif
}
__device__ __forceinline__ float frsq_(float x) {
#if defined(__HIP_DEVICE_COMPILE__)
  return __builtin_amdgcn_rsqf(x);
#else
  return 1.0f / sqrtf(x);
#endif
}
__device__ __forceinline__ float fexp_(float x)    { return fexp2_(x * LOG2E); }
__device__ __forceinline__ float fsigmoid(float x) { return frcp_(1.0f + fexp2_(-x * LOG2E)); }
// tanh(x) = 1 - 2/(e^{2x}+1): exact saturation at +/-inf, no branches
__device__ __forceinline__ float ftanh(float x)    { return 1.0f - 2.0f * frcp_(1.0f + fexp2_(x * (2.0f * LOG2E))); }

// ---------------------------------------------------------------------------
// WMMA helpers (CDNA5 gfx1250, wave32).  Layouts per ISA 7.12.2:
//   A (16x32 bf16): lane L -> row m0+(L&15); element j -> k = 16*(j/8) + 8*(L>>4) + (j%8)
//   B (32x16 bf16): lane L -> col n0+(L&15); element j -> k = 16*(j/8) + 8*(L>>4) + (j%8)
//   C/D (16x16 f32): lane L, vgpr r -> row m0 + 8*(L>>4) + r, col n0 + (L&15)
// ---------------------------------------------------------------------------

__device__ __forceinline__ v16bf load_afrag(const bf16* A, int lda, int m0, int k0, int lane) {
  const int m  = m0 + (lane & 15);
  const int kh = lane >> 4;
  const bf16* p = A + (size_t)m * lda + k0 + kh * 8;
  v8bf lo = *(const v8bf*)p;          // k = k0 + kh*8 + [0..7]
  v8bf hi = *(const v8bf*)(p + 16);   // k = k0 + 16 + kh*8 + [0..7]
  v16bf r;
#pragma unroll
  for (int i = 0; i < 8; ++i) { r[i] = lo[i]; r[i + 8] = hi[i]; }
  return r;
}

// B[k][n] = W[n][k] for W row-major (N x ldw): computes act @ W^T
__device__ __forceinline__ v16bf load_bfrag(const bf16* W, int ldw, int n0, int k0, int lane) {
  const int n  = n0 + (lane & 15);
  const int kh = lane >> 4;
  const bf16* p = W + (size_t)n * ldw + k0 + kh * 8;    // k = kh*8+[0..7], 16+kh*8+[0..7]
  v8bf lo = *(const v8bf*)p;
  v8bf hi = *(const v8bf*)(p + 16);
  v16bf r;
#pragma unroll
  for (int i = 0; i < 8; ++i) { r[i] = lo[i]; r[i + 8] = hi[i]; }
  return r;
}

__device__ __forceinline__ v8f wmma_bf16(v16bf a, v16bf b, v8f c) {
#if defined(__HIP_DEVICE_COMPILE__)
  return __builtin_amdgcn_wmma_f32_16x16x32_bf16(false, a, false, b, (short)0, c, false, false);
#else
  return c;
#endif
}

// ---------------------------------------------------------------------------
// Tensor Data Mover: DMA a 2D bf16 tile (BT rows x 32 cols, row stride S*32
// elements) from global into LDS.  D# built per ISA ch.8 (count=1, type=2,
// data_size=2B).  Issued by one wave; completion via s_wait_tensorcnt.
// ---------------------------------------------------------------------------
#if defined(USE_TDM)
__device__ __forceinline__ void tdm_load_tile(const bf16* gsrc, unsigned lds_off, int S) {
  unsigned long long ga = (unsigned long long)(size_t)gsrc;
  u32x4 g0;
  g0[0] = 1u;                                              // count=1 (valid), user mode
  g0[1] = lds_off;                                         // lds_addr (bytes)
  g0[2] = (unsigned)(ga & 0xffffffffu);                    // global_addr[31:0]
  g0[3] = (unsigned)((ga >> 32) & 0x01ffffffu) | (2u << 30); // addr[56:32] | type=2
  unsigned long long stride = (unsigned long long)S * 32u; // elements
  i32x8 g1;
  g1[0] = (int)(1u << 16);                 // data_size=1 -> 2 bytes/elem
  g1[1] = (int)(32u << 16);                // tensor_dim0 = 32 (lo16)
  g1[2] = (int)(32u << 16);                // tensor_dim0 hi=0 | tensor_dim1 lo = 32
  g1[3] = (int)(32u << 16);                // tensor_dim1 hi=0 | tile_dim0 = 32
  g1[4] = (int)(32u);                      // tile_dim1 = 32 | tile_dim2 = 0
  g1[5] = (int)(unsigned)(stride & 0xffffffffu);           // tensor_dim0_stride lo
  g1[6] = (int)(unsigned)((stride >> 32) & 0xffffu);       // stride hi | dim1_stride lo = 0
  g1[7] = 0;
  i32x4 z4 = {0, 0, 0, 0};
#if __clang_major__ >= 23
  i32x8 z8 = {0, 0, 0, 0, 0, 0, 0, 0};
  __builtin_amdgcn_tensor_load_to_lds(g0, g1, z4, z4, z8, 0);
#else
  __builtin_amdgcn_tensor_load_to_lds(g0, g1, z4, z4, 0);
#endif
}
#endif

// ---------------------------------------------------------------------------
// fp32 -> bf16 prepack (weights + x tensors)
// ---------------------------------------------------------------------------
__global__ void cvt_bf16_kernel(const float* __restrict__ src, bf16* __restrict__ dst, int n) {
  int i = blockIdx.x * blockDim.x + threadIdx.x;
  if (i < n) dst[i] = (bf16)src[i];
}

// ---------------------------------------------------------------------------
// LTC encoder: one workgroup = 32 batch rows, full sequence, 2 cells per step.
// State h lives in registers (each wave owns a 16x16 tile for the whole
// sequence); LDS holds only bf16 GEMM operands + the LN reduction buffer.
// ---------------------------------------------------------------------------
struct CellArg {
  const bf16 *Wg, *Win, *Wrec;              // bf16 (prepacked), row-major
  const float *tau, *bg, *lng, *lnb;        // fp32
};
struct EncTArgs {
  CellArg c0, c1;
  const bf16* x;    // (B, S, 32) bf16 prepacked
  float* enc;       // (B, 6, 128)
  int S, t;
};

template <int INK>
__device__ __forceinline__ void run_cell(
    const CellArg& cw, const bf16* xin, float (&hreg)[8], bf16* hbf,
    bf16 (*s_hh)[BT * 64], float* s_hn, float* s_ms, float* s_rs,
    float bgn, float it, float lngn, float lnbn,
    int tid, int lane, int m0, int n0, int kh, int nc)
{
  const int ldg = INK + 64;
  float Gx[8], Xin[8], ksum[8], kprev[8], hhreg[8];

  // step-invariant input projections: Gx = x @ Wg_x^T, Xin = x @ Win^T
  v8f aGx = {0.f,0.f,0.f,0.f,0.f,0.f,0.f,0.f};
  v8f aXi = {0.f,0.f,0.f,0.f,0.f,0.f,0.f,0.f};
#pragma unroll
  for (int k0 = 0; k0 < INK; k0 += 32) {
    v16bf a = load_afrag(xin, INK, m0, k0, lane);
    aGx = wmma_bf16(a, load_bfrag(cw.Wg,  ldg, n0, k0, lane), aGx);
    aXi = wmma_bf16(a, load_bfrag(cw.Win, INK, n0, k0, lane), aXi);
  }
#pragma unroll
  for (int r = 0; r < 8; ++r) { Gx[r] = aGx[r]; Xin[r] = aXi[r]; ksum[r] = 0.f; kprev[r] = 0.f; }

  const float coef[4] = {0.f, 0.5f, 0.5f, 1.f};
  const float wgt[4]  = {1.f, 2.f, 2.f, 1.f};

#pragma unroll
  for (int sub = 0; sub < 4; ++sub) {
    const bf16* asrc;
    if (sub == 0) {
      // hh == h : bf16 copy of h already lives in hbf (written by previous
      // step's LN stage, behind a barrier) -> no store, no barrier.
#pragma unroll
      for (int r = 0; r < 8; ++r) hhreg[r] = hreg[r];
      asrc = hbf;
    } else {
      bf16* cur = s_hh[sub & 1];
#pragma unroll
      for (int r = 0; r < 8; ++r) {
        hhreg[r] = hreg[r] + coef[sub] * kprev[r];
        cur[(m0 + kh * 8 + r) * 64 + nc] = (bf16)hhreg[r];
      }
      __syncthreads();
      asrc = cur;
    }
    v8f aG = {0.f,0.f,0.f,0.f,0.f,0.f,0.f,0.f};
    v8f aR = {0.f,0.f,0.f,0.f,0.f,0.f,0.f,0.f};
#pragma unroll
    for (int k0 = 0; k0 < 64; k0 += 32) {
      v16bf a = load_afrag(asrc, 64, m0, k0, lane);
      aG = wmma_bf16(a, load_bfrag(cw.Wg,   ldg, n0, INK + k0, lane), aG);
      aR = wmma_bf16(a, load_bfrag(cw.Wrec, 64,  n0, k0,       lane), aR);
    }
#pragma unroll
    for (int r = 0; r < 8; ++r) {
      float g  = fsigmoid(ftanh(Gx[r] + aG[r] + bgn));
      float kc = -hhreg[r] * it + Xin[r] + g * aR[r];
      ksum[r] += wgt[sub] * kc;
      kprev[r] = kc;
    }
  }

  // hn = h + (k1+2k2+2k3+k4)/6 ; then LayerNorm over 64 dims + tanh
  float hn[8];
#pragma unroll
  for (int r = 0; r < 8; ++r) {
    hn[r] = hreg[r] + ksum[r] * (1.0f / 6.0f);
    s_hn[(m0 + kh * 8 + r) * 64 + nc] = hn[r];
  }
  __syncthreads();
  if (tid < BT) {
    const f4* p = (const f4*)(s_hn + tid * 64);
    float sm = 0.f, s2 = 0.f;
#pragma unroll
    for (int c = 0; c < 16; ++c) {
      f4 v = p[c];
      sm += v.x + v.y + v.z + v.w;
      s2 += v.x * v.x + v.y * v.y + v.z * v.z + v.w * v.w;
    }
    float mean = sm * (1.0f / 64.0f);
    float var  = s2 * (1.0f / 64.0f) - mean * mean;
    s_ms[tid] = mean;
    s_rs[tid] = frsq_(var + 1e-5f);
  }
  __syncthreads();
#pragma unroll
  for (int r = 0; r < 8; ++r) {
    const int row = m0 + kh * 8 + r;
    float v = ftanh((hn[r] - s_ms[row]) * s_rs[row] * lngn + lnbn);
    hreg[r] = v;
    hbf[row * 64 + nc] = (bf16)v;
  }
  __syncthreads();
}

__global__ __launch_bounds__(THREADS) void enc_kernel(EncTArgs args) {
  const int bbase = blockIdx.x * BT;
  const int tid  = threadIdx.x;
  const int lane = tid & 31;
  const int wid  = tid >> 5;
  const int m0 = (wid >> 2) << 4;   // 2 M-tiles
  const int n0 = (wid & 3) << 4;    // 4 N-tiles
  const int kh = lane >> 4;
  const int nc = n0 + (lane & 15);
  const int S  = args.S;

  __shared__ bf16  s_x[2][BT * 32];       // double-buffered x tile (TDM target)
  __shared__ bf16  s_h0bf[BT * 64];       // persistent bf16 state, cell 0
  __shared__ bf16  s_h1bf[BT * 64];       // persistent bf16 state, cell 1
  __shared__ bf16  s_hh[2][BT * 64];      // RK4 intermediate states (ping-pong)
  __shared__ float s_hn[BT * 64];         // LN reduction buffer
  __shared__ float s_ms[BT], s_rs[BT];

  // per-lane, per-cell constants (hoisted out of the time loop)
  const float it0  = 1.0f / (log1pf(expf(args.c0.tau[nc])) + 1.0f);
  const float bg0  = args.c0.bg[nc];
  const float lng0 = args.c0.lng[nc];
  const float lnb0 = args.c0.lnb[nc];
  const float it1  = 1.0f / (log1pf(expf(args.c1.tau[nc])) + 1.0f);
  const float bg1  = args.c1.bg[nc];
  const float lng1 = args.c1.lng[nc];
  const float lnb1 = args.c1.lnb[nc];

  float h0reg[8], h1reg[8];
#pragma unroll
  for (int r = 0; r < 8; ++r) { h0reg[r] = 0.f; h1reg[r] = 0.f; }
  for (int i = tid; i < BT * 64; i += THREADS) {
    s_h0bf[i] = (bf16)0.f; s_h1bf[i] = (bf16)0.f;
  }
  __syncthreads();

  const bf16* xg = args.x + (size_t)bbase * S * 32;

#if defined(USE_TDM)
  if (wid == 0) tdm_load_tile(xg, (unsigned)(size_t)(void*)&s_x[0][0], S);
#endif

  for (int s = 0; s < S; ++s) {
    bf16* xcur = s_x[s & 1];
#if defined(USE_TDM)
    if (wid == 0) {
      __builtin_amdgcn_s_wait_tensorcnt(0);    // xcur ready
      if (s + 1 < S)                            // prefetch next step's tile
        tdm_load_tile(xg + (size_t)(s + 1) * 32,
                      (unsigned)(size_t)(void*)&s_x[(s + 1) & 1][0], S);
    }
#else
    for (int i = tid; i < BT * 32; i += THREADS)
      xcur[i] = xg[(size_t)(i >> 5) * S * 32 + (size_t)s * 32 + (i & 31)];
#endif
    __syncthreads();
    run_cell<32>(args.c0, xcur, h0reg, s_h0bf, s_hh, s_hn, s_ms, s_rs,
                 bg0, it0, lng0, lnb0, tid, lane, m0, n0, kh, nc);
    run_cell<64>(args.c1, s_h0bf, h1reg, s_h1bf, s_hh, s_hn, s_ms, s_rs,
                 bg1, it1, lng1, lnb1, tid, lane, m0, n0, kh, nc);
  }

  // enc[b][t] = [h1, (h0+h1)/2] straight from registers
#pragma unroll
  for (int r = 0; r < 8; ++r) {
    const int row = m0 + kh * 8 + r;
    float* dst = args.enc + ((size_t)(bbase + row) * NTF + args.t) * 128;
    dst[nc]      = h1reg[r];
    dst[64 + nc] = 0.5f * (h0reg[r] + h1reg[r]);
  }
}

// ---------------------------------------------------------------------------
// Head: cross-timeframe attention + gating + pooling + fusion + classifier.
// One 128-thread block per sample; ~1% of total FLOPs.
// ---------------------------------------------------------------------------
struct HeadArgs {
  const float *enc, *ctx;
  const float *emb, *Wq, *bq, *Wk, *bk, *Wv, *bv, *Wo, *bo, *ag, *ab;
  const float *Wp, *bp, *g1, *b1, *Wg2, *bg2, *g2, *b2;
  const float *tW1, *tb1, *tW2, *tb2;
  const float *fW, *fb, *fg, *fb2;
  const float *cW1, *cb1, *cW2, *cb2;
  float* out;
};

__global__ __launch_bounds__(128) void head_kernel(HeadArgs a) {
  const int b   = blockIdx.x;
  const int tid = threadIdx.x;
  __shared__ float sx[768], sq[768], sk[768], sv[768], se[768];
  __shared__ float ssc[144], sglob[128], sfu[128], sfz[64], sh[192];
  __shared__ float smean[6], srstd[6], ssct[6];

  // x = enc + emb
  for (int i = tid; i < 768; i += 128) sx[i] = a.enc[(size_t)b * 768 + i] + a.emb[i];
  __syncthreads();
  // q,k,v projections
  for (int i = tid; i < 768; i += 128) {
    int t = i >> 7, j = i & 127;
    const float* xr = sx + t * 128;
    float aq = a.bq[j], ak = a.bk[j], av = a.bv[j];
#pragma unroll 4
    for (int c = 0; c < 128; ++c) {
      float xv = xr[c];
      aq += xv * a.Wq[j * 128 + c];
      ak += xv * a.Wk[j * 128 + c];
      av += xv * a.Wv[j * 128 + c];
    }
    sq[i] = aq; sk[i] = ak; sv[i] = av;
  }
  __syncthreads();
  // attention scores (4 heads x 6 x 6)
  for (int i = tid; i < 144; i += 128) {
    int h = i / 36, r = i % 36, tq = r / 6, tu = r % 6;
    float s = 0.f;
#pragma unroll
    for (int d = 0; d < 32; ++d) s += sq[tq * 128 + h * 32 + d] * sk[tu * 128 + h * 32 + d];
    ssc[i] = s * 0.17677669529663687f;   // 1/sqrt(32)
  }
  __syncthreads();
  if (tid < 24) {
    float* row = ssc + (tid / 6) * 36 + (tid % 6) * 6;
    float mx = row[0];
    for (int u = 1; u < 6; ++u) mx = fmaxf(mx, row[u]);
    float sum = 0.f;
    for (int u = 0; u < 6; ++u) { row[u] = fexp_(row[u] - mx); sum += row[u]; }
    float inv = frcp_(sum);
    for (int u = 0; u < 6; ++u) row[u] *= inv;
  }
  __syncthreads();
  // attention output
  for (int i = tid; i < 768; i += 128) {
    int t = i >> 7, j = i & 127, h = j >> 5;
    const float* w = ssc + h * 36 + t * 6;
    float s = 0.f;
#pragma unroll
    for (int u = 0; u < 6; ++u) s += w[u] * sv[u * 128 + j];
    se[i] = s;
  }
  __syncthreads();
  // Wo projection -> sx
  for (int i = tid; i < 768; i += 128) {
    int t = i >> 7, j = i & 127;
    const float* xr = se + t * 128;
    float s = a.bo[j];
#pragma unroll 4
    for (int c = 0; c < 128; ++c) s += xr[c] * a.Wo[j * 128 + c];
    sx[i] = s;
  }
  __syncthreads();
  // LN + relu -> se (= enc after attention)
  if (tid < 6) {
    float s = 0.f, s2 = 0.f;
    for (int c = 0; c < 128; ++c) { float v = sx[tid * 128 + c]; s += v; s2 += v * v; }
    float m = s * (1.f / 128.f);
    smean[tid] = m; srstd[tid] = frsq_(s2 * (1.f / 128.f) - m * m + 1e-5f);
  }
  __syncthreads();
  for (int i = tid; i < 768; i += 128) {
    int t = i >> 7, j = i & 127;
    se[i] = fmaxf((sx[i] - smean[t]) * srstd[t] * a.ag[j] + a.ab[j], 0.f);
  }
  __syncthreads();
  // global pathway: glob = relu(LN(flat(se) @ Wp^T + bp))
  if (tid < 128) {
    float s = a.bp[tid];
    const float* wr = a.Wp + (size_t)tid * 768;
#pragma unroll 4
    for (int c = 0; c < 768; ++c) s += se[c] * wr[c];
    sglob[tid] = s;
  }
  __syncthreads();
  if (tid == 0) {
    float s = 0.f, s2 = 0.f;
    for (int c = 0; c < 128; ++c) { float v = sglob[c]; s += v; s2 += v * v; }
    float m = s * (1.f / 128.f);
    smean[0] = m; srstd[0] = frsq_(s2 * (1.f / 128.f) - m * m + 1e-5f);
  }
  __syncthreads();
  if (tid < 128) sglob[tid] = fmaxf((sglob[tid] - smean[0]) * srstd[0] * a.g1[tid] + a.b1[tid], 0.f);
  __syncthreads();
  // gate: sigmoid(LN([se, glob] @ Wg^T + bg)) -> blend
  for (int i = tid; i < 768; i += 128) {
    int t = i >> 7, j = i & 127;
    const float* wr = a.Wg2 + (size_t)j * 256;
    float s = a.bg2[j];
#pragma unroll 4
    for (int c = 0; c < 128; ++c) s += se[t * 128 + c] * wr[c];
#pragma unroll 4
    for (int c = 0; c < 128; ++c) s += sglob[c] * wr[128 + c];
    sx[i] = s;
  }
  __syncthreads();
  if (tid < 6) {
    float s = 0.f, s2 = 0.f;
    for (int c = 0; c < 128; ++c) { float v = sx[tid * 128 + c]; s += v; s2 += v * v; }
    float m = s * (1.f / 128.f);
    smean[tid] = m; srstd[tid] = frsq_(s2 * (1.f / 128.f) - m * m + 1e-5f);
  }
  __syncthreads();
  for (int i = tid; i < 768; i += 128) {
    int t = i >> 7, j = i & 127;
    float g = fsigmoid((sx[i] - smean[t]) * srstd[t] * a.g2[j] + a.b2[j]);
    se[i] = se[i] * g + sglob[j] * (1.f - g);
  }
  __syncthreads();
  // adaptive timeframe attention pooling
  for (int i = tid; i < 192; i += 128) {
    int t = i / 32, kk = i % 32;
    float s = a.tb1[kk];
#pragma unroll 4
    for (int c = 0; c < 128; ++c) s += se[t * 128 + c] * a.tW1[kk * 128 + c];
    sh[i] = ftanh(s);
  }
  __syncthreads();
  if (tid < 6) {
    float s = a.tb2[0];
#pragma unroll
    for (int kk = 0; kk < 32; ++kk) s += a.tW2[kk] * sh[tid * 32 + kk];
    ssct[tid] = s;
  }
  __syncthreads();
  if (tid == 0) {
    float mx = ssct[0];
    for (int t = 1; t < 6; ++t) mx = fmaxf(mx, ssct[t]);
    float sum = 0.f;
    for (int t = 0; t < 6; ++t) { ssct[t] = fexp_(ssct[t] - mx); sum += ssct[t]; }
    float inv = frcp_(sum);
    for (int t = 0; t < 6; ++t) ssct[t] *= inv;
  }
  __syncthreads();
  if (tid < 128) {
    float s = 0.f;
#pragma unroll
    for (int t = 0; t < 6; ++t) s += ssct[t] * se[t * 128 + tid];
    sfu[tid] = s;
  }
  __syncthreads();
  // fusion with context
  if (tid < 64) {
    const float* wr = a.fW + (size_t)tid * 160;
    float s = a.fb[tid];
#pragma unroll 4
    for (int c = 0; c < 128; ++c) s += sfu[c] * wr[c];
#pragma unroll
    for (int c = 0; c < 32; ++c) s += a.ctx[(size_t)b * 32 + c] * wr[128 + c];
    sfz[tid] = s;
  }
  __syncthreads();
  if (tid == 0) {
    float s = 0.f, s2 = 0.f;
    for (int c = 0; c < 64; ++c) { float v = sfz[c]; s += v; s2 += v * v; }
    float m = s * (1.f / 64.f);
    smean[0] = m; srstd[0] = frsq_(s2 * (1.f / 64.f) - m * m + 1e-5f);
  }
  __syncthreads();
  if (tid < 64) sfz[tid] = fmaxf((sfz[tid] - smean[0]) * srstd[0] * a.fg[tid] + a.fb2[tid], 0.f);
  __syncthreads();
  // classifier
  if (tid < 32) {
    float s = a.cb1[tid];
#pragma unroll
    for (int c = 0; c < 64; ++c) s += sfz[c] * a.cW1[tid * 64 + c];
    sh[tid] = fmaxf(s, 0.f);
  }
  __syncthreads();
  if (tid == 0) {
    float s = a.cb2[0];
#pragma unroll
    for (int kk = 0; kk < 32; ++kk) s += a.cW2[kk] * sh[kk];
    a.out[b] = s;
  }
}

// ---------------------------------------------------------------------------
// host launcher
// ---------------------------------------------------------------------------
extern "C" void kernel_launch(void* const* d_in, const int* in_sizes, int n_in,
                              void* d_out, int out_size, void* d_ws, size_t ws_size,
                              hipStream_t stream) {
  (void)n_in; (void)ws_size; (void)out_size;
  static const int SEQ[NTF] = {96, 72, 64, 48, 36, 24};
  const int B = in_sizes[6] / 32;   // context is (B, 32)

  // Input order: x tensors, context, then params flattened in insertion order.
  const float* xs[NTF];
  for (int i = 0; i < NTF; ++i) xs[i] = (const float*)d_in[i];
  const float* ctx = (const float*)d_in[6];

  // enc params: 6 timeframes x 2 layers x {tau, Win, Wrec, Wg, bg, ln_g, ln_b}
  const float* ptau[NTF][2]; const float* pWin[NTF][2]; const float* pWrec[NTF][2];
  const float* pWg[NTF][2];  const float* pbg[NTF][2];
  const float* plng[NTF][2]; const float* plnb[NTF][2];
  int pi = 7;
  for (int t = 0; t < NTF; ++t)
    for (int l = 0; l < 2; ++l) {
      ptau[t][l] = (const float*)d_in[pi + 0];
      pWin[t][l] = (const float*)d_in[pi + 1];
      pWrec[t][l] = (const float*)d_in[pi + 2];
      pWg[t][l]  = (const float*)d_in[pi + 3];
      pbg[t][l]  = (const float*)d_in[pi + 4];
      plng[t][l] = (const float*)d_in[pi + 5];
      plnb[t][l] = (const float*)d_in[pi + 6];
      pi += 7;
    }

  // workspace: [ enc fp32 (B,6,128) | bf16 weights | bf16 x tensors ]
  float* enc = (float*)d_ws;
  bf16* wbase = (bf16*)(enc + (size_t)B * NTF * 128);

  bf16* bWg[NTF][2]; bf16* bWin[NTF][2]; bf16* bWrec[NTF][2];
  size_t woff = 0;
  for (int t = 0; t < NTF; ++t)
    for (int l = 0; l < 2; ++l) {
      const int in = (l == 0) ? 32 : 64;
      int nWg = 64 * (in + 64), nWin = 64 * in, nWrec = 64 * 64;
      bWg[t][l] = wbase + woff;
      cvt_bf16_kernel<<<(nWg + 255) / 256, 256, 0, stream>>>(pWg[t][l], bWg[t][l], nWg);
      woff += (size_t)nWg;
      bWin[t][l] = wbase + woff;
      cvt_bf16_kernel<<<(nWin + 255) / 256, 256, 0, stream>>>(pWin[t][l], bWin[t][l], nWin);
      woff += (size_t)nWin;
      bWrec[t][l] = wbase + woff;
      cvt_bf16_kernel<<<(nWrec + 255) / 256, 256, 0, stream>>>(pWrec[t][l], bWrec[t][l], nWrec);
      woff += (size_t)nWrec;
    }
  bf16* xbf[NTF];
  for (int t = 0; t < NTF; ++t) {
    int n = B * SEQ[t] * 32;
    xbf[t] = wbase + woff;
    cvt_bf16_kernel<<<(n + 255) / 256, 256, 0, stream>>>(xs[t], xbf[t], n);
    woff += (size_t)n;
  }

  // encoders: one launch per timeframe (no dynamic kernarg indexing)
  for (int t = 0; t < NTF; ++t) {
    EncTArgs ea;
    ea.c0 = {bWg[t][0], bWin[t][0], bWrec[t][0], ptau[t][0], pbg[t][0], plng[t][0], plnb[t][0]};
    ea.c1 = {bWg[t][1], bWin[t][1], bWrec[t][1], ptau[t][1], pbg[t][1], plng[t][1], plnb[t][1]};
    ea.x = xbf[t];
    ea.enc = enc;
    ea.S = SEQ[t];
    ea.t = t;
    enc_kernel<<<dim3(B / BT), dim3(THREADS), 0, stream>>>(ea);
  }

  // head
  HeadArgs ha;
  ha.enc = enc; ha.ctx = ctx;
  ha.emb = (const float*)d_in[91];
  ha.Wq = (const float*)d_in[92];  ha.bq = (const float*)d_in[93];
  ha.Wk = (const float*)d_in[94];  ha.bk = (const float*)d_in[95];
  ha.Wv = (const float*)d_in[96];  ha.bv = (const float*)d_in[97];
  ha.Wo = (const float*)d_in[98];  ha.bo = (const float*)d_in[99];
  ha.ag = (const float*)d_in[100]; ha.ab = (const float*)d_in[101];
  ha.Wp = (const float*)d_in[102]; ha.bp = (const float*)d_in[103];
  ha.g1 = (const float*)d_in[104]; ha.b1 = (const float*)d_in[105];
  ha.Wg2 = (const float*)d_in[106]; ha.bg2 = (const float*)d_in[107];
  ha.g2 = (const float*)d_in[108]; ha.b2 = (const float*)d_in[109];
  ha.tW1 = (const float*)d_in[110]; ha.tb1 = (const float*)d_in[111];
  ha.tW2 = (const float*)d_in[112]; ha.tb2 = (const float*)d_in[113];
  ha.fW = (const float*)d_in[114]; ha.fb = (const float*)d_in[115];
  ha.fg = (const float*)d_in[116]; ha.fb2 = (const float*)d_in[117];
  ha.cW1 = (const float*)d_in[118]; ha.cb1 = (const float*)d_in[119];
  ha.cW2 = (const float*)d_in[120]; ha.cb2 = (const float*)d_in[121];
  ha.out = (float*)d_out;
  head_kernel<<<dim3(B), dim3(128), 0, stream>>>(ha);
}
